// DecoderRNN_46634754900667
// MI455X (gfx1250) — matrix operand order repaired and checked by reading, not compiled, not run on hardware
//
#include <hip/hip_runtime.h>
#include <hip/hip_bf16.h>

typedef __attribute__((ext_vector_type(16))) __bf16 v16bf;
typedef __attribute__((ext_vector_type(8)))  __bf16 v8bf;
typedef __attribute__((ext_vector_type(4)))  __bf16 v4bf;
typedef __attribute__((ext_vector_type(8)))  float  v8f;
typedef __attribute__((ext_vector_type(4)))  float  v4f;

namespace {
constexpr int kE = 256;   // embedding dim
constexpr int kH = 512;   // hidden dim
constexpr int kT = 512;   // timesteps
constexpr int kB = 128;   // batch
constexpr int kK = 512;   // GEMM K depth (2E == H)
}

// ---------------------------------------------------------------- WMMA core
__device__ __forceinline__ v8f wmma_bf16(v16bf a, v16bf b, v8f c) {
  // v_wmma_f32_16x16x32_bf16: D = A(16x32) * B(32x16) + C, fp32 accumulate
  return __builtin_amdgcn_wmma_f32_16x16x32_bf16(
      /*neg_a=*/false, a, /*neg_b=*/false, b,
      /*c_mod=*/(short)0, c, /*reuse_a=*/false, /*reuse_b=*/false);
}

__device__ __forceinline__ v16bf make_frag(v8bf lo, v8bf hi) {
  v16bf r;
#pragma unroll
  for (int i = 0; i < 8; ++i) { r[i] = lo[i]; r[i + 8] = hi[i]; }
  return r;
}

// A fragment (16x32, bf16) from row-major [rows][kK] in LDS.
// Lane l: row = mT + (l&15); elements 0..7 hold K = k0+8h.., 8..15 hold K = k0+16+8h..
__device__ __forceinline__ v16bf frag_a(const __bf16* A, int mT, int k0, int lane) {
  const __bf16* p = A + (mT + (lane & 15)) * kK + k0 + 8 * (lane >> 4);
  return make_frag(*(const v8bf*)p, *(const v8bf*)(p + 16));
}

// B fragment (32x16) from row-major W[N][K] in global (bf16).
// Lane l: col n = nT + (l&15); elements 0..7 hold K = k0+8h.., 8..15 hold K = k0+16+8h..
__device__ __forceinline__ v16bf frag_b(const __bf16* W, int nT, int K, int k0, int lane) {
  const __bf16* p = W + (size_t)(nT + (lane & 15)) * K + k0 + 8 * (lane >> 4);
  return make_frag(*(const v8bf*)p, *(const v8bf*)(p + 16));
}

// ---------------------------------------------------------------- activations
__device__ __forceinline__ float sigm(float x) {
  return 1.0f / (1.0f + __expf(-x));
}
__device__ __forceinline__ float tanh_f(float x) {
  // 1 - 2/(e^{2x}+1): saturates cleanly to +/-1, no inf/inf NaN
  float e = __expf(2.0f * x);
  return 1.0f - 2.0f / (e + 1.0f);
}

// ---------------------------------------------------------------- LSTM layer
// gates = A @ W^T (+bih+bhh); h = sigm(i)*... ; f-gate skipped (c_prev == 0).
// A: LDS [128][512] bf16. W: global bf16 [2048][512]. Hout: LDS [128][512] bf16.
__device__ __forceinline__ void lstm_layer(const __bf16* __restrict__ A,
                                           const __bf16* __restrict__ W,
                                           const float* __restrict__ bih,
                                           const float* __restrict__ bhh,
                                           __bf16* __restrict__ Hout,
                                           int lane, int wave) {
  // 8 M-tiles x 32 N-tiles = 256 h-tiles; 32 waves -> 8 tiles each.
#pragma unroll 1
  for (int i = 0; i < 8; ++i) {
    int tile = wave + 32 * i;
    int mT = (tile >> 5) << 4;   // 0..112
    int nT = (tile & 31) << 4;   // 0..496 (h column base)
    v8f ci = {}, cg = {}, co = {};
#pragma unroll 4
    for (int k0 = 0; k0 < kK; k0 += 32) {
      v16bf a  = frag_a(A, mT, k0, lane);           // shared across 3 gates
      v16bf bi = frag_b(W, nT,        kK, k0, lane); // i-gate rows
      v16bf bg = frag_b(W, nT + 1024, kK, k0, lane); // g-gate rows
      v16bf bo = frag_b(W, nT + 1536, kK, k0, lane); // o-gate rows
      ci = wmma_bf16(a, bi, ci);
      cg = wmma_bf16(a, bg, cg);
      co = wmma_bf16(a, bo, co);
    }
    int n = nT + (lane & 15);
    float Bi = bih[n] + bhh[n];
    float Bg = bih[n + 1024] + bhh[n + 1024];
    float Bo = bih[n + 1536] + bhh[n + 1536];
    int mBase = mT + 8 * (lane >> 4);
#pragma unroll
    for (int v = 0; v < 8; ++v) {
      float cc = sigm(ci[v] + Bi) * tanh_f(cg[v] + Bg);
      float hh = sigm(co[v] + Bo) * tanh_f(cc);
      Hout[(mBase + v) * kK + n] = (__bf16)hh;
    }
  }
}

// ---------------------------------------------------------------- args
struct SeqArgs {
  const float*  feats[2];
  const __bf16* w0[2];
  const __bf16* w1[2];
  const float*  bih0[2];
  const float*  bhh0[2];
  const float*  bih1[2];
  const float*  bhh1[2];
  const __bf16* wl;     // lin_W bf16 [256][512]
  const float*  linb;   // lin_b  [256]
  float*        out;    // [2][T*B][E]
};

// ---------------------------------------------------------------- persistent scan
__global__ __launch_bounds__(1024) void decoder_scan(SeqArgs args) {
  __shared__ __align__(16) __bf16 bufX[kB * kK]; // x (then h1)  : 128 KB
  __shared__ __align__(16) __bf16 bufH[kB * kH]; // h0           : 128 KB

  const int br   = blockIdx.x;
  const int tid  = threadIdx.x;
  const int lane = tid & 31;
  const int wave = tid >> 5;

  const float*  feats = args.feats[br];
  const __bf16* w0    = args.w0[br];
  const __bf16* w1    = args.w1[br];
  const float*  bih0  = args.bih0[br];
  const float*  bhh0  = args.bhh0[br];
  const float*  bih1  = args.bih1[br];
  const float*  bhh1  = args.bhh1[br];
  float* out = args.out + (size_t)br * kT * kB * kE;

  // t==0: prev_y is zero -> zero the y-half of x.  (4096 chunks of 8 bf16)
  for (int c = tid; c < kB * (kE / 8); c += 1024) {
    int b = c >> 5, e0 = (c & 31) * 8;
    v8bf z;
#pragma unroll
    for (int i = 0; i < 8; ++i) z[i] = (__bf16)0.0f;
    *(v8bf*)&bufX[b * kK + e0] = z;
  }

  for (int t = 0; t < kT; ++t) {
    // ---- stage 0: feat_t -> bf16 -> bufX[:, 256:512]
#pragma unroll
    for (int j = 0; j < 4; ++j) {
      int c = tid + j * 1024;          // 0..4095
      int b = c >> 5;
      int e0 = (c & 31) * 8;
      const float* fp = feats + ((size_t)b * kT + t) * kE + e0;
      v4f f0 = *(const v4f*)fp;
      v4f f1 = *(const v4f*)(fp + 4);
      v8bf fb;
#pragma unroll
      for (int i = 0; i < 4; ++i) { fb[i] = (__bf16)f0[i]; fb[i + 4] = (__bf16)f1[i]; }
      *(v8bf*)&bufX[b * kK + kE + e0] = fb;
    }
    __syncthreads();

    // ---- stage 1: layer-1 LSTM cell: bufX -> bufH
    lstm_layer(bufX, w0, bih0, bhh0, bufH, lane, wave);
    __syncthreads();

    // ---- stage 2: layer-2 LSTM cell: bufH -> bufX (h1 overwrites x)
    lstm_layer(bufH, w1, bih1, bhh1, bufX, lane, wave);
    __syncthreads();

    // ---- stage 3: linear y = h1 @ lin_W^T + b ; store fp32 out, keep regs
    v8f yv[4];
    int mTs[4], nTs[4];
#pragma unroll 1
    for (int i = 0; i < 4; ++i) {
      int tile = wave + 32 * i;        // 0..127  (8 M-tiles x 16 N-tiles)
      int mT = (tile >> 4) << 4;
      int nT = (tile & 15) << 4;
      mTs[i] = mT; nTs[i] = nT;
      v8f acc = {};
#pragma unroll 4
      for (int k0 = 0; k0 < kK; k0 += 32) {
        v16bf a  = frag_a(bufX, mT, k0, lane);
        v16bf bw = frag_b(args.wl, nT, kK, k0, lane);
        acc = wmma_bf16(a, bw, acc);
      }
      int n = nT + (lane & 15);
      float bb = args.linb[n];
      int mBase = mT + 8 * (lane >> 4);
#pragma unroll
      for (int v = 0; v < 8; ++v) {
        float y = acc[v] + bb;
        out[((size_t)t * kB + mBase + v) * kE + n] = y;  // t-major: [T*B, E]
        acc[v] = y;
      }
      yv[i] = acc;
    }
    __syncthreads();   // linear reads of bufX complete before feedback write

    // ---- stage 4: feedback y -> bufX[:, 0:256] for step t+1
#pragma unroll
    for (int i = 0; i < 4; ++i) {
      int n = nTs[i] + (lane & 15);
      int mBase = mTs[i] + 8 * (lane >> 4);
#pragma unroll
      for (int v = 0; v < 8; ++v)
        bufX[(mBase + v) * kK + n] = (__bf16)yv[i][v];
    }
    // next iteration's stage-0 writes the disjoint feat half; the sync after
    // stage 0 orders both against the next layer-1 reads.
  }
}

// ---------------------------------------------------------------- fp32 -> bf16
__global__ void cvt_bf16_kernel(const float* __restrict__ src,
                                __bf16* __restrict__ dst, int n4) {
  int i = blockIdx.x * blockDim.x + threadIdx.x;
  if (i < n4) {
    v4f f = *(const v4f*)(src + 4 * (size_t)i);
    v4bf o;
#pragma unroll
    for (int j = 0; j < 4; ++j) o[j] = (__bf16)f[j];
    *(v4bf*)(dst + 4 * (size_t)i) = o;
  }
}

// ---------------------------------------------------------------- launch
extern "C" void kernel_launch(void* const* d_in, const int* in_sizes, int n_in,
                              void* d_out, int out_size, void* d_ws, size_t ws_size,
                              hipStream_t stream) {
  (void)in_sizes; (void)n_in; (void)out_size; (void)ws_size;

  const float* up_feat  = (const float*)d_in[0];
  const float* lo_feat  = (const float*)d_in[1];
  const float* upW0     = (const float*)d_in[2];
  const float* upBih0   = (const float*)d_in[3];
  const float* upBhh0   = (const float*)d_in[4];
  const float* upW1     = (const float*)d_in[5];
  const float* upBih1   = (const float*)d_in[6];
  const float* upBhh1   = (const float*)d_in[7];
  const float* loW0     = (const float*)d_in[8];
  const float* loBih0   = (const float*)d_in[9];
  const float* loBhh0   = (const float*)d_in[10];
  const float* loW1     = (const float*)d_in[11];
  const float* loBih1   = (const float*)d_in[12];
  const float* loBhh1   = (const float*)d_in[13];
  const float* linW     = (const float*)d_in[14];
  const float* linB     = (const float*)d_in[15];

  // workspace: bf16 weight copies (~8.6 MB)
  constexpr size_t nGate = 2048ull * 512ull;   // Wih0 / Wih1 elements
  constexpr size_t nLin  = 256ull * 512ull;    // lin_W elements
  char* ws = (char*)d_ws;
  __bf16* w0u = (__bf16*)ws;                ws += nGate * 2;
  __bf16* w1u = (__bf16*)ws;                ws += nGate * 2;
  __bf16* w0l = (__bf16*)ws;                ws += nGate * 2;
  __bf16* w1l = (__bf16*)ws;                ws += nGate * 2;
  __bf16* wl  = (__bf16*)ws;                ws += nLin * 2;

  const int thr = 256;
  int g4 = (int)(nGate / 4), l4 = (int)(nLin / 4);
  cvt_bf16_kernel<<<(g4 + thr - 1) / thr, thr, 0, stream>>>(upW0, w0u, g4);
  cvt_bf16_kernel<<<(g4 + thr - 1) / thr, thr, 0, stream>>>(upW1, w1u, g4);
  cvt_bf16_kernel<<<(g4 + thr - 1) / thr, thr, 0, stream>>>(loW0, w0l, g4);
  cvt_bf16_kernel<<<(g4 + thr - 1) / thr, thr, 0, stream>>>(loW1, w1l, g4);
  cvt_bf16_kernel<<<(l4 + thr - 1) / thr, thr, 0, stream>>>(linW, wl,  l4);

  SeqArgs a;
  a.feats[0] = up_feat;  a.feats[1] = lo_feat;
  a.w0[0] = w0u;  a.w0[1] = w0l;
  a.w1[0] = w1u;  a.w1[1] = w1l;
  a.bih0[0] = upBih0; a.bih0[1] = loBih0;
  a.bhh0[0] = upBhh0; a.bhh0[1] = loBhh0;
  a.bih1[0] = upBih1; a.bih1[1] = loBih1;
  a.bhh1[0] = upBhh1; a.bhh1[1] = loBhh1;
  a.wl = wl; a.linb = linB;
  a.out = (float*)d_out;

  decoder_scan<<<2, 1024, 0, stream>>>(a);
}
